// MTFM_3470333575206
// MI455X (gfx1250) — compile-verified, matrix operand order checked by reading, not run on hardware
//
#include <hip/hip_runtime.h>
#include <hip/hip_bf16.h>

typedef __attribute__((ext_vector_type(16))) _Float16 v16h;
typedef __attribute__((ext_vector_type(8)))  _Float16 v8h;
typedef __attribute__((ext_vector_type(8)))  float    v8f;

#define BB    256
#define LL    200
#define LPAD  240
#define DD    300
#define DPAD  320
#define KC    256
#define AA    1000
#define APAD  1008
#define FF    256
#define CC    100
#define CPAD  112
#define E_DIM 1024
#define CAT_LD 2016
#define UMMF_LD 1024

// ---------------------------------------------------------------------------
// CDNA5 WMMA fragment loader: 16-bit A/B fragment (16x32 tile of a row-major
// (rows x K) matrix).  lane<16: row=lane, K={0..7,16..23}; lane>=16: same row,
// K={8..15,24..31}.  Two 16B loads (global or LDS depending on pointer).
// ---------------------------------------------------------------------------
__device__ __forceinline__ v16h load_frag(const _Float16* base, int ld) {
  int lane = threadIdx.x & 31;
  int row  = lane & 15;
  int koff = (lane >> 4) << 3;           // 0 or 8
  const _Float16* p = base + (long)row * ld + koff;
  v8h lo = *reinterpret_cast<const v8h*>(p);
  v8h hi = *reinterpret_cast<const v8h*>(p + 16);
  v16h f;
#pragma unroll
  for (int i = 0; i < 8; ++i) { f[i] = lo[i]; f[8 + i] = hi[i]; }
  return f;
}

__device__ __forceinline__ v8f wmma_f16(v16h a, v16h b, v8f c) {
  return __builtin_amdgcn_wmma_f32_16x16x32_f16(false, a, false, b,
                                                (short)0, c, false, false);
}

// gfx1250 async global -> LDS copy, 16B per lane (ASYNCcnt tracked)
__device__ __forceinline__ void async_g2l_b128(unsigned lds_off,
                                               const _Float16* gp) {
  unsigned long long ga = (unsigned long long)(uintptr_t)gp;
  asm volatile("global_load_async_to_lds_b128 %0, %1, off"
               :: "v"(lds_off), "v"(ga) : "memory");
}
__device__ __forceinline__ void wait_async0() {
  asm volatile("s_wait_asynccnt 0" ::: "memory");
}

// ---------------------------------------------------------------------------
// Prep kernels
// ---------------------------------------------------------------------------
__global__ void k_zero_u32(unsigned* p, long n) {
  long i = (long)blockIdx.x * blockDim.x + threadIdx.x;
  if (i < n) p[i] = 0u;
}

// x[b*LPAD + l][DPAD] f16 = emb_table[idx[b,l]][d], zero padded
__global__ void k_build_x(const int* __restrict__ idx,
                          const float* __restrict__ emb,
                          _Float16* __restrict__ x) {
  int row = blockIdx.x;                   // b*LPAD + l
  int b = row / LPAD, l = row % LPAD;
  _Float16* xr = x + (long)row * DPAD;
  if (l >= LL) {
    for (int d = threadIdx.x; d < DPAD; d += blockDim.x) xr[d] = (_Float16)0.f;
    return;
  }
  int t = idx[b * LL + l];
  const float* er = emb + (long)t * DD;
  for (int d = threadIdx.x; d < DPAD; d += blockDim.x)
    xr[d] = (d < DD) ? (_Float16)er[d] : (_Float16)0.f;
}

// conv weights -> wpack[slice(14)][k(256)][DPAD] f16, slice = (h_idx, shift j)
__global__ void k_pack_convw(const float* __restrict__ w0, const float* __restrict__ w1,
                             const float* __restrict__ w2, const float* __restrict__ w3,
                             _Float16* __restrict__ wp) {
  int r = blockIdx.x;                     // s*256 + k
  int s = r >> 8, k = r & 255;
  int hi, j;
  if      (s < 2) { hi = 0; j = s;     }
  else if (s < 5) { hi = 1; j = s - 2; }
  else if (s < 9) { hi = 2; j = s - 5; }
  else            { hi = 3; j = s - 9; }
  const float* w = (hi == 0) ? w0 : (hi == 1) ? w1 : (hi == 2) ? w2 : w3;
  int h = hi + 2;
  _Float16* o = wp + (long)r * DPAD;
  for (int d = threadIdx.x; d < DPAD; d += blockDim.x)
    o[d] = (d < DD) ? (_Float16)w[(long)k * DD * h + (long)d * h + j] : (_Float16)0.f;
}

// (N x Kin) f32 row-major -> (Npad x Kpad) f16, zero padded
__global__ void k_pack_mat(const float* __restrict__ src, _Float16* __restrict__ dst,
                           int N, int Kin, int Npad, int Kpad) {
  long i = (long)blockIdx.x * blockDim.x + threadIdx.x;
  long total = (long)Npad * Kpad;
  if (i >= total) return;
  int n = (int)(i / Kpad), k = (int)(i % Kpad);
  dst[i] = (n < N && k < Kin) ? (_Float16)src[(long)n * Kin + k] : (_Float16)0.f;
}

// api_emb (F x A) -> apiT (APAD x F) f16
__global__ void k_pack_apiT(const float* __restrict__ src, _Float16* __restrict__ dst) {
  long i = (long)blockIdx.x * blockDim.x + threadIdx.x;
  if (i >= (long)APAD * FF) return;
  int a = (int)(i / FF), f = (int)(i % FF);
  dst[i] = (a < AA) ? (_Float16)src[(long)f * AA + a] : (_Float16)0.f;
}

// ---------------------------------------------------------------------------
// Fused conv(2,3,4,5) + bias + relu + maxpool.
// Block = 128 threads / 4 waves, handles one (b, ptile-pair):
//   1) async-stage 36x320 f16 x-tile into LDS (global_load_async_to_lds_b128)
//   2) each wave computes 4 ktiles; per (kk, shift j): 2 A frags (LDS, the two
//      position tiles) and up to 4 weight frags (global), each weight frag
//      feeding two WMMAs -> 28 WMMAs per k-step, 8 f32 accumulators.
//   3) bias + relu + position mask + pool via shfl + u32 atomicMax.
// ---------------------------------------------------------------------------
__global__ __launch_bounds__(128) void k_conv_pool(
    const _Float16* __restrict__ x, const _Float16* __restrict__ wpack,
    const float* __restrict__ b0, const float* __restrict__ b1,
    const float* __restrict__ b2, const float* __restrict__ b3,
    unsigned* __restrict__ epool) {
  __shared__ __align__(16) _Float16 xt[36 * DPAD];   // 23 KB
  int pt0 = blockIdx.x * 2;               // ptile pair: pt0, pt0+1 (0..13)
  int b   = blockIdx.y;                   // 0..255
  const int hbase[4] = {0, 2, 5, 9};

  // ---- async stage x rows [pt0*16, pt0*16+35] into LDS ----
  const _Float16* xb = x + ((long)b * LPAD + (long)pt0 * 16) * DPAD;
  unsigned lds_base = (unsigned)(uintptr_t)&xt[0];
  for (int c = threadIdx.x; c < 1440; c += 128) {    // 36 rows * 40 16B chunks
    int r = c / 40, col = (c % 40) * 8;
    async_g2l_b128(lds_base + (unsigned)c * 16, xb + (long)r * DPAD + col);
  }
  wait_async0();
  __syncthreads();

  int wave = threadIdx.x >> 5;
  int lane = threadIdx.x & 31;
  const float* bptr[4] = {b0, b1, b2, b3};

  for (int kt = 0; kt < 4; ++kt) {
    int ktile = wave * 4 + kt;            // 0..15
    v8f acc[2][4] = {{v8f{}, v8f{}, v8f{}, v8f{}},
                     {v8f{}, v8f{}, v8f{}, v8f{}}};

    for (int kk = 0; kk < DPAD; kk += 32) {
#pragma unroll
      for (int j = 0; j < 5; ++j) {
        v16h a0 = load_frag(&xt[j * DPAD + kk], DPAD);        // ptile pt0
        v16h a1 = load_frag(&xt[(16 + j) * DPAD + kk], DPAD); // ptile pt0+1
#pragma unroll
        for (int hi = 0; hi < 4; ++hi) {
          if (j < hi + 2) {
            v16h w = load_frag(
                wpack + ((long)(hbase[hi] + j) * KC + (long)ktile * 16) * DPAD + kk,
                DPAD);
            acc[0][hi] = wmma_f16(a0, w, acc[0][hi]);
            acc[1][hi] = wmma_f16(a1, w, acc[1][hi]);
          }
        }
      }
    }

    int colk = ktile * 16 + (lane & 15);
#pragma unroll
    for (int pt = 0; pt < 2; ++pt) {
      int ptile = pt0 + pt;
      int rbase = ptile * 16 + ((lane >> 4) ? 8 : 0);
#pragma unroll
      for (int hi = 0; hi < 4; ++hi) {
        float bias = bptr[hi][colk];
        int h = hi + 2;
        float m = 0.f;
#pragma unroll
        for (int r = 0; r < 8; ++r) {
          int p = rbase + r;
          float v = acc[pt][hi][r] + bias;
          v = v > 0.f ? v : 0.f;          // relu
          if (p > LL - h) v = 0.f;        // mask invalid positions
          m = fmaxf(m, v);
        }
        float o = __shfl_xor(m, 16, 32);  // combine row halves (same col)
        m = fmaxf(m, o);
        if (lane < 16)
          atomicMax(&epool[(long)b * E_DIM + colk * 4 + hi], __float_as_uint(m));
      }
    }
  }
}

__global__ void k_e_convert(const unsigned* __restrict__ ep, _Float16* __restrict__ e) {
  long i = (long)blockIdx.x * blockDim.x + threadIdx.x;
  if (i < (long)BB * E_DIM) e[i] = (_Float16)__uint_as_float(ep[i]);
}

// ---------------------------------------------------------------------------
// Generic WMMA GEMM:  C = act(A(MxK,f16) * Bt(NxK,f16)^T + bias)
// block = 4 waves, each wave one 16x16 output tile.
// act: 0=none 1=tanh 2=sigmoid.  Optional f16 dest (ldch) and f32 dest (clip ncf).
// ---------------------------------------------------------------------------
__global__ __launch_bounds__(128) void k_gemm(
    const _Float16* __restrict__ A, int lda,
    const _Float16* __restrict__ Bt, int ldb,
    const float* __restrict__ bias, int nbias,
    _Float16* __restrict__ Ch, long ldch,
    float* __restrict__ Cf, long ldcf, int ncf,
    int Npad, int Kin, int act) {
  int wave  = threadIdx.x >> 5;
  int ntile = blockIdx.x * 4 + wave;
  int mtile = blockIdx.y;
  if (ntile * 16 >= Npad) return;         // wave-uniform

  const _Float16* Ap = A  + (long)mtile * 16 * lda;
  const _Float16* Bp = Bt + (long)ntile * 16 * ldb;
  v8f acc = v8f{};
  for (int kk = 0; kk < Kin; kk += 32) {
    v16h a = load_frag(Ap + kk, lda);
    v16h b = load_frag(Bp + kk, ldb);
    acc = wmma_f16(a, b, acc);
  }

  int lane = threadIdx.x & 31;
  int col  = ntile * 16 + (lane & 15);
  int row0 = mtile * 16 + ((lane >> 4) ? 8 : 0);
  float bs = (bias != nullptr && col < nbias) ? bias[col] : 0.f;
#pragma unroll
  for (int r = 0; r < 8; ++r) {
    int row = row0 + r;
    float v = acc[r] + bs;
    if (act == 1)      v = tanhf(v);
    else if (act == 2) v = 1.f / (1.f + __expf(-v));
    if (Ch) Ch[(long)row * ldch + col] = (_Float16)v;
    if (Cf && col < ncf) Cf[(long)row * ldcf + col] = v;
  }
}

// ---------------------------------------------------------------------------
// Collapsed MLP:  u_mlp[b,a] = tanh(s[b] + t[a] + c)
// ---------------------------------------------------------------------------
__global__ void k_mlp_vectors(const float* __restrict__ w1, const float* __restrict__ w2,
                              const float* __restrict__ b1, const float* __restrict__ b2,
                              float* __restrict__ va, float* __restrict__ vb,
                              float* __restrict__ cbuf) {
  int f = threadIdx.x;
  if (f < FF) {
    float sa = 0.f, sb = 0.f;
    for (int fp = 0; fp < FF; ++fp) {
      float v2 = w2[fp];
      sa += v2 * w1[(long)fp * (2 * FF) + f];
      sb += v2 * w1[(long)fp * (2 * FF) + FF + f];
    }
    va[f] = sa; vb[f] = sb;
  }
  if (f == 0) {
    float c = b2[0];
    for (int g = 0; g < FF; ++g) c += b1[g] * w2[g];
    cbuf[0] = c;
  }
}

__global__ void k_mlp_s(const _Float16* __restrict__ ust, const float* __restrict__ va,
                        float* __restrict__ s) {
  int b = threadIdx.x;                    // 256 threads
  float acc = 0.f;
  for (int f = 0; f < FF; ++f) acc += (float)ust[(long)b * FF + f] * va[f];
  s[b] = acc;
}

__global__ void k_mlp_t(const float* __restrict__ api_emb, const float* __restrict__ vb,
                        float* __restrict__ t) {
  int a = blockIdx.x * blockDim.x + threadIdx.x;
  if (a >= AA) return;
  float acc = 0.f;
  for (int f = 0; f < FF; ++f) acc += api_emb[(long)f * AA + a] * vb[f];
  t[a] = acc;
}

__global__ void k_umlp(const float* __restrict__ s, const float* __restrict__ t,
                       const float* __restrict__ cbuf, _Float16* __restrict__ cat1) {
  long i = (long)blockIdx.x * blockDim.x + threadIdx.x;
  if (i >= (long)BB * AA) return;
  int b = (int)(i / AA), a = (int)(i % AA);
  cat1[(long)b * CAT_LD + AA + a] = (_Float16)tanhf(s[b] + t[a] + cbuf[0]);
}

// ---------------------------------------------------------------------------
// Launch
// ---------------------------------------------------------------------------
static inline size_t al256(size_t x) { return (x + 255) & ~(size_t)255; }

extern "C" void kernel_launch(void* const* d_in, const int* in_sizes, int n_in,
                              void* d_out, int out_size, void* d_ws, size_t ws_size,
                              hipStream_t stream) {
  const int*   idx      = (const int*)  d_in[0];
  const float* emb      = (const float*)d_in[1];
  const float* cw[4]    = {(const float*)d_in[2], (const float*)d_in[4],
                           (const float*)d_in[6], (const float*)d_in[8]};
  const float* cb[4]    = {(const float*)d_in[3], (const float*)d_in[5],
                           (const float*)d_in[7], (const float*)d_in[9]};
  const float* sc_w     = (const float*)d_in[10];
  const float* sc_b     = (const float*)d_in[11];
  const float* ficfc_w  = (const float*)d_in[12];
  const float* ficfc_b  = (const float*)d_in[13];
  const float* api_emb  = (const float*)d_in[14];
  const float* mlp_w1   = (const float*)d_in[15];
  const float* mlp_b1   = (const float*)d_in[16];
  const float* mlp_w2   = (const float*)d_in[17];
  const float* mlp_b2   = (const float*)d_in[18];
  const float* ficfcl_w = (const float*)d_in[19];
  const float* ficfcl_b = (const float*)d_in[20];
  const float* fusion_w = (const float*)d_in[21];
  const float* fusion_b = (const float*)d_in[22];
  const float* api_w    = (const float*)d_in[23];
  const float* api_b    = (const float*)d_in[24];
  const float* cat_w    = (const float*)d_in[25];
  const float* cat_b    = (const float*)d_in[26];
  float* out_y = (float*)d_out;                 // 256 x 1000
  float* out_z = (float*)d_out + (long)BB * AA; // 256 x 100

  // ---- workspace carve (~60 MB) ----
  char* ws = (char*)d_ws; size_t cur = 0;
  auto take = [&](size_t bytes) { void* p = ws + cur; cur = al256(cur + bytes); return p; };
  _Float16* x        = (_Float16*)take((size_t)BB * LPAD * DPAD * 2);
  _Float16* wpack    = (_Float16*)take((size_t)14 * KC * DPAD * 2);
  unsigned* epool    = (unsigned*)take((size_t)BB * E_DIM * 4);
  _Float16* e16      = (_Float16*)take((size_t)BB * E_DIM * 2);
  _Float16* sc_p     = (_Float16*)take((size_t)APAD * E_DIM * 2);
  _Float16* ficfc_p  = (_Float16*)take((size_t)FF * E_DIM * 2);
  _Float16* apiT     = (_Float16*)take((size_t)APAD * FF * 2);
  _Float16* ficfcl_p = (_Float16*)take((size_t)APAD * CAT_LD * 2);
  _Float16* fusion_p = (_Float16*)take((size_t)APAD * CAT_LD * 2);
  _Float16* apiw_p   = (_Float16*)take((size_t)APAD * UMMF_LD * 2);
  _Float16* catw_p   = (_Float16*)take((size_t)CPAD * UMMF_LD * 2);
  _Float16* ust      = (_Float16*)take((size_t)BB * FF * 2);
  _Float16* cat1     = (_Float16*)take((size_t)BB * CAT_LD * 2);
  _Float16* cat2     = (_Float16*)take((size_t)BB * CAT_LD * 2);
  _Float16* ummf     = (_Float16*)take((size_t)BB * UMMF_LD * 2);
  float*    va       = (float*)take(FF * 4);
  float*    vb       = (float*)take(FF * 4);
  float*    sbuf     = (float*)take(BB * 4);
  float*    tbuf     = (float*)take(APAD * 4);
  float*    cbuf     = (float*)take(4);
  (void)ws_size; (void)in_sizes; (void)n_in; (void)out_size;

  auto zero = [&](void* p, long n_u32) {
    k_zero_u32<<<dim3((unsigned)((n_u32 + 255) / 256)), 256, 0, stream>>>((unsigned*)p, n_u32);
  };
  // deterministic re-init each call
  zero(epool, (long)BB * E_DIM);
  zero(cat1,  (long)BB * CAT_LD / 2);
  zero(cat2,  (long)BB * CAT_LD / 2);
  zero(ummf,  (long)BB * UMMF_LD / 2);

  // ---- prep / pack ----
  k_build_x<<<BB * LPAD, 256, 0, stream>>>(idx, emb, x);
  k_pack_convw<<<14 * KC, 256, 0, stream>>>(cw[0], cw[1], cw[2], cw[3], wpack);
  auto pack = [&](const float* s, _Float16* d, int N, int Kin, int Np, int Kp) {
    long tot = (long)Np * Kp;
    k_pack_mat<<<dim3((unsigned)((tot + 255) / 256)), 256, 0, stream>>>(s, d, N, Kin, Np, Kp);
  };
  pack(sc_w,     sc_p,     AA, E_DIM, APAD, E_DIM);
  pack(ficfc_w,  ficfc_p,  FF, E_DIM, FF,   E_DIM);
  pack(ficfcl_w, ficfcl_p, AA, 2 * AA, APAD, CAT_LD);
  pack(fusion_w, fusion_p, AA, 2 * AA, APAD, CAT_LD);
  pack(api_w,    apiw_p,   AA, AA,    APAD, UMMF_LD);
  pack(cat_w,    catw_p,   CC, AA,    CPAD, UMMF_LD);
  k_pack_apiT<<<dim3((APAD * FF + 255) / 256), 256, 0, stream>>>(api_emb, apiT);
  k_mlp_vectors<<<1, 256, 0, stream>>>(mlp_w1, mlp_w2, mlp_b1, mlp_b2, va, vb, cbuf);
  k_mlp_t<<<4, 256, 0, stream>>>(api_emb, vb, tbuf);

  // ---- conv + pool (dominant ~110 GFLOP, v_wmma + async LDS staging) ----
  k_conv_pool<<<dim3(7, BB), 128, 0, stream>>>(x, wpack, cb[0], cb[1], cb[2], cb[3], epool);
  k_e_convert<<<dim3((BB * E_DIM + 255) / 256), 256, 0, stream>>>(epool, e16);

  auto gemm = [&](const _Float16* A, int lda, const _Float16* Bt, int ldb,
                  const float* bias, int nbias, _Float16* Ch, long ldch,
                  float* Cf, long ldcf, int ncf, int Npad, int Kin, int act) {
    dim3 grid((unsigned)((Npad + 63) / 64), BB / 16);
    k_gemm<<<grid, 128, 0, stream>>>(A, lda, Bt, ldb, bias, nbias,
                                     Ch, ldch, Cf, ldcf, ncf, Npad, Kin, act);
  };

  // u_sc -> cat2[:, 0:1008]
  gemm(e16, E_DIM, sc_p, E_DIM, sc_b, AA, cat2, CAT_LD, nullptr, 0, 0, APAD, E_DIM, 0);
  // u_sc_trans (ust)
  gemm(e16, E_DIM, ficfc_p, E_DIM, ficfc_b, FF, ust, FF, nullptr, 0, 0, FF, E_DIM, 0);
  // s[b] from ust
  k_mlp_s<<<1, 256, 0, stream>>>(ust, va, sbuf);
  // u_mm -> cat1[:, 0:1008]
  gemm(ust, FF, apiT, FF, nullptr, 0, cat1, CAT_LD, nullptr, 0, 0, APAD, FF, 0);
  // u_mlp -> cat1[:, 1000:2000]  (overwrites the 1000..1007 pad from u_mm)
  k_umlp<<<dim3((BB * AA + 255) / 256), 256, 0, stream>>>(sbuf, tbuf, cbuf, cat1);
  // u_fic = tanh(cat1 @ fic_fcl^T) -> cat2[:, 1000:2008]
  gemm(cat1, CAT_LD, ficfcl_p, CAT_LD, ficfcl_b, AA, cat2 + AA, CAT_LD,
       nullptr, 0, 0, APAD, CAT_LD, 1);
  // u_mmf = cat2 @ fusion^T -> ummf (ld 1024, cols 1008..1023 stay zero)
  gemm(cat2, CAT_LD, fusion_p, CAT_LD, fusion_b, AA, ummf, UMMF_LD,
       nullptr, 0, 0, APAD, CAT_LD, 0);
  // y = sigmoid(ummf @ api_w^T) -> d_out[0 : 256*1000]
  gemm(ummf, UMMF_LD, apiw_p, UMMF_LD, api_b, AA, nullptr, 0,
       out_y, AA, AA, APAD, UMMF_LD, 2);
  // z = sigmoid(ummf @ cat_w^T) -> d_out[256*1000 : +256*100]
  gemm(ummf, UMMF_LD, catw_p, UMMF_LD, cat_b, CC, nullptr, 0,
       out_z, CC, CC, CPAD, UMMF_LD, 2);
}
